// SparseAdaHyperedgeGen_25099788878229
// MI455X (gfx1250) — compile-verified, matrix operand order checked by reading, not run on hardware
//
#include <hip/hip_runtime.h>
#include <math.h>

// ---------------------------------------------------------------------------
// SparseAdaHyperedgeGen for MI455X (gfx1250, wave32, WMMA)
//   B=8, N=4096, D=512, E=128, H=8 -> scale 1/(H*sqrt(D/H)) = 1/64, k=64
// Key facts used:
//  * einsum('bnhd,behd->bne') sums over h AND d == full-D dot: logits = Xp P^T / 64
//  * Wc GEMM (268MB weights, M=8) is bandwidth bound -> wave-per-row dot product
//  * Xp GEMM (17.2 GFLOP) + logits GEMM (4.3 GFLOP) -> v_wmma_f32_16x16x32_bf16
// ---------------------------------------------------------------------------

#define B_ 8
#define N_ 4096
#define D_ 512
#define E_ 128
#define TOPK_ 64
#define M_TOT (B_ * N_)   // 32768 rows

typedef __attribute__((ext_vector_type(16))) __bf16 v16bf;
typedef __attribute__((ext_vector_type(8)))  float  v8f;

// ---- WMMA fragment loaders (ISA 7.12.2, 16-bit A 16x32 layout) -------------
// lane<16: M=lane,   K in {0..7, 16..23}
// lane>=16: M=lane-16, K in {8..15, 24..31}
__device__ __forceinline__ v16bf load_frag_bf16(const __bf16* src, int ld, int lane) {
  v16bf f;
  const __bf16* row = src + (size_t)(lane & 15) * ld + ((lane & 16) ? 8 : 0);
#pragma unroll
  for (int i = 0; i < 8; ++i) f[i] = row[i];
#pragma unroll
  for (int i = 0; i < 8; ++i) f[8 + i] = row[16 + i];
  return f;
}

__device__ __forceinline__ v16bf load_frag_f32(const float* src, int ld, int lane) {
  v16bf f;
  const float* row = src + (size_t)(lane & 15) * ld + ((lane & 16) ? 8 : 0);
#pragma unroll
  for (int i = 0; i < 8; ++i) f[i] = (__bf16)row[i];
#pragma unroll
  for (int i = 0; i < 8; ++i) f[8 + i] = (__bf16)row[16 + i];
  return f;
}

// ---- Stage 1: ctx = [mean, max] over N -------------------------------------
__global__ void ctx_partial_kernel(const float* __restrict__ X,
                                   float* __restrict__ psum,
                                   float* __restrict__ pmax) {
  // grid: B*8*4 blocks of 128 threads; each thread one d-column, 512 n's
  int bid = blockIdx.x;
  int b  = bid >> 5;
  int nc = (bid >> 2) & 7;
  int dc = bid & 3;
  int d  = dc * 128 + threadIdx.x;
  const float* base = X + ((size_t)b * N_ + (size_t)nc * 512) * D_ + d;
  float s = 0.f, m = -__builtin_inff();
  for (int n = 0; n < 512; ++n) {
    float v = base[(size_t)n * D_];
    s += v;
    m = fmaxf(m, v);
  }
  psum[(b * 8 + nc) * D_ + d] = s;
  pmax[(b * 8 + nc) * D_ + d] = m;
}

__global__ void ctx_final_kernel(const float* __restrict__ psum,
                                 const float* __restrict__ pmax,
                                 float* __restrict__ ctx) {
  int i = blockIdx.x * blockDim.x + threadIdx.x;   // 0..4095
  int b = i >> 9, d = i & 511;
  float s = 0.f, m = -__builtin_inff();
#pragma unroll
  for (int c = 0; c < 8; ++c) {
    s += psum[(b * 8 + c) * D_ + d];
    m = fmaxf(m, pmax[(b * 8 + c) * D_ + d]);
  }
  ctx[b * 1024 + d]       = s * (1.0f / (float)N_);
  ctx[b * 1024 + 512 + d] = m;
}

// ---- Stage 2a: Wp -> bf16 ---------------------------------------------------
__global__ void cvt_wp_kernel(const float* __restrict__ Wp, __bf16* __restrict__ WpB) {
  int i = blockIdx.x * 256 + threadIdx.x;
  WpB[i] = (__bf16)Wp[i];
}

// ---- Stage 2b: prototypes[b, e, d] = proto_base + bc + ctx[b] . Wc_row -----
// Bandwidth bound on Wc (268 MB): one wave per Wc row, row read once and
// applied to all 8 batch ctx vectors staged in LDS.
__global__ void proto_kernel(const float* __restrict__ Wc,
                             const float* __restrict__ bc,
                             const float* __restrict__ proto_base,
                             const float* __restrict__ ctx,
                             __bf16* __restrict__ protoB) {
  __shared__ float sctx[B_ * 1024];
  for (int i = threadIdx.x; i < B_ * 1024; i += 256) sctx[i] = ctx[i];
  __syncthreads();

  int lane = threadIdx.x & 31;
  int row  = blockIdx.x * 8 + (threadIdx.x >> 5);   // row = e*D + d, 0..65535
  const float* w = Wc + (size_t)row * 1024;
  float acc[B_] = {0.f, 0.f, 0.f, 0.f, 0.f, 0.f, 0.f, 0.f};
  for (int c = lane; c < 1024; c += 32) {
    __builtin_prefetch(w + c + 512, 0, 1);          // global_prefetch_b8
    float wv = w[c];
#pragma unroll
    for (int bb = 0; bb < B_; ++bb)
      acc[bb] = fmaf(wv, sctx[bb * 1024 + c], acc[bb]);
  }
#pragma unroll
  for (int bb = 0; bb < B_; ++bb) {
#pragma unroll
    for (int off = 16; off > 0; off >>= 1)
      acc[bb] += __shfl_xor(acc[bb], off, 32);
  }
  if (lane == 0) {
    float base = bc[row] + proto_base[row];
#pragma unroll
    for (int bb = 0; bb < B_; ++bb)
      protoB[(size_t)bb * (E_ * D_) + row] = (__bf16)(base + acc[bb]);
  }
}

// ---- Stage 3: Xp = X . Wp^T + bp  (bf16 WMMA, f32 accumulate) --------------
__global__ void xp_gemm_kernel(const float* __restrict__ X,
                               const __bf16* __restrict__ WpB,
                               const float* __restrict__ bp,
                               __bf16* __restrict__ Xp) {
  int lane = threadIdx.x & 31;
  int gid  = blockIdx.x * 8 + (threadIdx.x >> 5);   // 65536 wave tiles
  int mt = gid >> 5;                                // 2048 row tiles
  int nt = gid & 31;                                // 32 col tiles
  int m0 = mt * 16, n0 = nt * 16;

  const float*  a_base = X + (size_t)m0 * D_;
  const __bf16* b_base = WpB + (size_t)n0 * D_;     // row n of Wp == col n of B
  v8f c = {};
  for (int k0 = 0; k0 < D_; k0 += 32) {
    v16bf a = load_frag_f32(a_base + k0, D_, lane);
    v16bf b = load_frag_bf16(b_base + k0, D_, lane);
    c = __builtin_amdgcn_wmma_f32_16x16x32_bf16(false, a, false, b,
                                                (short)0, c, false, false);
  }
  // C layout: lane holds col n0+(lane&15), rows v + 8*(lane>=16)
  int col   = n0 + (lane & 15);
  float bias = bp[col];
  int rbase = m0 + ((lane & 16) ? 8 : 0);
#pragma unroll
  for (int v = 0; v < 8; ++v)
    Xp[(size_t)(rbase + v) * D_ + col] = (__bf16)(c[v] + bias);
}

// ---- Stage 4: logits[b,n,e] = Xp[b,n,:] . P[b,e,:] / 64  -------------------
// Block = 16 N-rows of one batch; A tile staged in LDS; 8 waves = 8 E-tiles.
__global__ void logits_kernel(const __bf16* __restrict__ Xp,
                              const __bf16* __restrict__ protoB,
                              float* __restrict__ logits) {
  __shared__ __bf16 tileA[16 * D_];                 // 16 KB
  int b  = blockIdx.x >> 8;
  int mt = blockIdx.x & 255;
  size_t m0 = (size_t)b * N_ + (size_t)mt * 16;

  const uint4* src = (const uint4*)(Xp + m0 * D_);
  uint4* dst = (uint4*)tileA;
  for (int i = threadIdx.x; i < (16 * D_) / 8; i += 256) dst[i] = src[i];
  __syncthreads();

  int lane = threadIdx.x & 31;
  int w    = threadIdx.x >> 5;                      // E-tile 0..7
  const __bf16* pb = protoB + (size_t)b * (E_ * D_) + (size_t)(w * 16) * D_;
  v8f c = {};
  for (int k0 = 0; k0 < D_; k0 += 32) {
    v16bf a = load_frag_bf16(tileA + k0, D_, lane); // ds_load path
    v16bf f = load_frag_bf16(pb + k0, D_, lane);
    c = __builtin_amdgcn_wmma_f32_16x16x32_bf16(false, a, false, f,
                                                (short)0, c, false, false);
  }
  const float scale = 1.0f / 64.0f;                 // 1/(H*sqrt(dh))
  int e     = w * 16 + (lane & 15);
  int rbase = mt * 16 + ((lane & 16) ? 8 : 0);
#pragma unroll
  for (int v = 0; v < 8; ++v)
    logits[((size_t)b * N_ + rbase + v) * E_ + e] = c[v] * scale;
}

// ---- Stage 5: per-node top-64 + softmax ------------------------------------
// One wave per row of 128 logits: 64 iterative argmax reductions on a
// monotonic (value, ~index) 64-bit key -> ties pick the lowest index,
// output is sorted descending (matches jax.lax.top_k).
__global__ void topk_kernel(const float* __restrict__ logits,
                            int* __restrict__ topi,
                            float* __restrict__ edge_w) {
  int lane = threadIdx.x & 31;
  int row  = blockIdx.x * 8 + (threadIdx.x >> 5);   // 0..32767
  const float* lrow = logits + (size_t)row * E_;

  float vals[4];
#pragma unroll
  for (int j = 0; j < 4; ++j) vals[j] = lrow[lane + 32 * j];

  float sel_v[2] = {0.f, 0.f};
  int   sel_i[2] = {0, 0};
  float vmax = 0.f;

  for (int t = 0; t < TOPK_; ++t) {
    unsigned long long best = 0ull;
#pragma unroll
    for (int j = 0; j < 4; ++j) {
      unsigned ub  = __float_as_uint(vals[j]);
      unsigned key = (ub & 0x80000000u) ? ~ub : (ub | 0x80000000u);
      unsigned idx = (unsigned)(lane + 32 * j);
      unsigned long long cand = ((unsigned long long)key << 32) | (unsigned)(~idx);
      best = (cand > best) ? cand : best;
    }
#pragma unroll
    for (int off = 16; off > 0; off >>= 1) {
      unsigned long long o = __shfl_xor(best, off, 32);
      best = (o > best) ? o : best;
    }
    unsigned key = (unsigned)(best >> 32);
    unsigned idx = ~(unsigned)best;
    unsigned ub  = (key & 0x80000000u) ? (key & 0x7fffffffu) : ~key;
    float v = __uint_as_float(ub);
    if (t == 0) vmax = v;
    if ((int)(idx & 31u) == lane) vals[idx >> 5] = -__builtin_inff();
    if ((t & 31) == lane) { sel_v[t >> 5] = v; sel_i[t >> 5] = (int)idx; }
  }

  float e0 = __expf(sel_v[0] - vmax);
  float e1 = __expf(sel_v[1] - vmax);
  float s = e0 + e1;
#pragma unroll
  for (int off = 16; off > 0; off >>= 1) s += __shfl_xor(s, off, 32);
  float inv = 1.0f / s;

  size_t o = (size_t)row * TOPK_;
  topi[o + lane]        = sel_i[0];
  topi[o + 32 + lane]   = sel_i[1];
  edge_w[o + lane]      = e0 * inv;
  edge_w[o + 32 + lane] = e1 * inv;
}

// ---------------------------------------------------------------------------
extern "C" void kernel_launch(void* const* d_in, const int* in_sizes, int n_in,
                              void* d_out, int out_size, void* d_ws, size_t ws_size,
                              hipStream_t stream) {
  const float* X     = (const float*)d_in[0];
  const float* proto = (const float*)d_in[1];
  const float* Wc    = (const float*)d_in[2];
  const float* bc    = (const float*)d_in[3];
  const float* Wp    = (const float*)d_in[4];
  const float* bp    = (const float*)d_in[5];

  char* ws = (char*)d_ws;
  size_t off = 0;
  float*  psum   = (float*)(ws + off);  off += (size_t)B_ * 8 * D_ * 4;      // 128 KB
  float*  pmax   = (float*)(ws + off);  off += (size_t)B_ * 8 * D_ * 4;      // 128 KB
  float*  ctx    = (float*)(ws + off);  off += (size_t)B_ * 1024 * 4;        // 32 KB
  __bf16* WpB    = (__bf16*)(ws + off); off += (size_t)D_ * D_ * 2;          // 512 KB
  __bf16* protoB = (__bf16*)(ws + off); off += (size_t)B_ * E_ * D_ * 2;     // 1 MB
  __bf16* XpB    = (__bf16*)(ws + off); off += (size_t)M_TOT * D_ * 2;       // 32 MB
  float*  logits = (float*)(ws + off);  off += (size_t)M_TOT * E_ * 4;       // 16 MB

  int*   topi = (int*)d_out;
  float* ew   = (float*)d_out + (size_t)M_TOT * TOPK_;

  ctx_partial_kernel<<<B_ * 8 * 4, 128, 0, stream>>>(X, psum, pmax);
  ctx_final_kernel<<<16, 256, 0, stream>>>(psum, pmax, ctx);
  cvt_wp_kernel<<<(D_ * D_) / 256, 256, 0, stream>>>(Wp, WpB);
  proto_kernel<<<(E_ * D_) / 8, 256, 0, stream>>>(Wc, bc, proto, ctx, protoB);
  xp_gemm_kernel<<<(M_TOT / 16) * (D_ / 16) / 8, 256, 0, stream>>>(X, WpB, bp, XpB);
  logits_kernel<<<B_ * (N_ / 16), 256, 0, stream>>>(XpB, protoB, logits);
  topk_kernel<<<M_TOT / 8, 256, 0, stream>>>(logits, topi, ew);
}
